// MultiHeadSelfAttention_26010321944713
// MI455X (gfx1250) — matrix operand order repaired and checked by reading, not compiled, not run on hardware
//
#include <hip/hip_runtime.h>
#include <hip/hip_bf16.h>

// ---------------------------------------------------------------------------
// MI455X (gfx1250) windowed multi-head attention, bf16 WMMA pipeline.
//  S=8192, E=1024, H=16, D=64, WINDOW=256.
// ---------------------------------------------------------------------------

typedef __bf16 v16bf __attribute__((ext_vector_type(16)));
typedef float  v8f   __attribute__((ext_vector_type(8)));
typedef unsigned int v4u __attribute__((ext_vector_type(4)));

#define WMMA_BF16(a, b, c) \
  __builtin_amdgcn_wmma_f32_16x16x32_bf16(false, (a), false, (b), (short)0, (c), false, false)

// --- CDNA5 async global->LDS path (guarded; falls back to ds_store staging) --
#if defined(__has_builtin)
#if __has_builtin(__builtin_amdgcn_global_load_async_to_lds_b128) && \
    __has_builtin(__builtin_amdgcn_s_wait_asynccnt)
#define HAVE_ASYNC 1
#endif
#endif

#ifdef HAVE_ASYNC
// Builtin signature (from hipcc diagnostic): param0 = v4i addrspace(1)*,
// param1 = v4i addrspace(3)*, then imm offset, imm cpol.
typedef int v4i_g __attribute__((vector_size(16)));
typedef __attribute__((address_space(1))) v4i_g* gas_p;
typedef __attribute__((address_space(3))) v4i_g* las_p;
#define ASYNC_CP(g, l) \
  __builtin_amdgcn_global_load_async_to_lds_b128((gas_p)(g), (las_p)(l), 0, 0)
#define ASYNC_WAIT() __builtin_amdgcn_s_wait_asynccnt(0)
#else
#define ASYNC_CP(g, l) (*(v4u*)(l) = *(const v4u*)(g))
#define ASYNC_WAIT()
#endif

union Frag16 {
  v16bf v;
  v4u q[2];
  unsigned short u[16];
};

__device__ __forceinline__ unsigned short f2bf(float x) {
  unsigned int u = __float_as_uint(x);
  unsigned int r = u + 0x7FFFu + ((u >> 16) & 1u);  // round-to-nearest-even
  return (unsigned short)(r >> 16);
}

// 16-bit A-matrix 16x32 (MxK) fragment, contiguous k order (matches B):
// lane L<16: M=L,    K = 0..15;  lane L>=16: M=L-16, K = 16..31
__device__ __forceinline__ v16bf load_a_frag(const unsigned short* __restrict__ base,
                                             int ld, int m0, int k0, int lane) {
  int M  = lane & 15;
  int kb = (lane & 16) ? 16 : 0;
  const unsigned short* p = base + (size_t)(m0 + M) * ld + k0 + kb;
  Frag16 f;
  f.q[0] = *reinterpret_cast<const v4u*>(p);
  f.q[1] = *reinterpret_cast<const v4u*>(p + 8);
  return f.v;
}

// 16-bit B-matrix 32x16 (KxN) fragment, B[k][n] = Row[n][k] (row-major over k):
// lane L<16: N=L, K=0..15 ; lane L>=16: N=L-16, K=16..31
__device__ __forceinline__ v16bf load_b_frag(const unsigned short* __restrict__ base,
                                             int ld, int n0, int k0, int lane) {
  int N  = lane & 15;
  int kb = (lane & 16) ? 16 : 0;
  const unsigned short* p = base + (size_t)(n0 + N) * ld + k0 + kb;
  Frag16 f;
  f.q[0] = *reinterpret_cast<const v4u*>(p);
  f.q[1] = *reinterpret_cast<const v4u*>(p + 8);
  return f.v;
}

// B fragment from an LDS row pointer (contiguous 16 elems at p)
__device__ __forceinline__ v16bf lds_b_frag(const unsigned short* p) {
  Frag16 f;
  f.q[0] = *reinterpret_cast<const v4u*>(p);
  f.q[1] = *reinterpret_cast<const v4u*>(p + 8);
  return f.v;
}

// ---------------------------------------------------------------------------
// Kernel 0: f32 -> bf16 conversion
// ---------------------------------------------------------------------------
__global__ void cvt_f32_bf16(const float* __restrict__ src,
                             unsigned short* __restrict__ dst, int n) {
  int i = blockIdx.x * blockDim.x + threadIdx.x;
  if (i < n) dst[i] = f2bf(src[i]);
}

// ---------------------------------------------------------------------------
// Kernel 1: fused QKV projection (x @ W^T) + per-head LayerNorm + RoPE.
// Grid: (S/64, H). Block: 192 threads = 6 waves.
//   wave = mat(0..2) * 2 + strip(0..1). Each wave: 32x64 strip, software-
//   pipelined (double-buffered fragments, unroll-by-2 over K).
// q,k stored [H][S][D] bf16; v stored transposed [H][D][S] bf16.
// ---------------------------------------------------------------------------
__global__ void __launch_bounds__(192)
qkv_ln_rope(const unsigned short* __restrict__ xb,
            const unsigned short* __restrict__ Wqb,
            const unsigned short* __restrict__ Wkb,
            const unsigned short* __restrict__ Wvb,
            const float* __restrict__ qg, const float* __restrict__ kg,
            unsigned short* __restrict__ qbuf,
            unsigned short* __restrict__ kbuf,
            unsigned short* __restrict__ vT) {
  const int S = 8192, E = 1024, D = 64;
  __shared__ __align__(16) float tile[3][64][65];

  const int rb = blockIdx.x, h = blockIdx.y;
  const int tid = threadIdx.x, lane = tid & 31, wave = tid >> 5;
  const int mat = wave >> 1, strip = wave & 1;
  const unsigned short* Wb = (mat == 0) ? Wqb : ((mat == 1) ? Wkb : Wvb);
  const int m0 = rb * 64 + strip * 32;
  const int n0 = h * 64;

  v8f acc[2][4] = {};
  v16bf a0[2], b0[4], a1[2], b1[4];
#pragma unroll
  for (int s = 0; s < 2; ++s) a0[s] = load_a_frag(xb, E, m0 + s * 16, 0, lane);
#pragma unroll
  for (int t = 0; t < 4; ++t) b0[t] = load_b_frag(Wb, E, n0 + t * 16, 0, lane);

  for (int k0 = 0; k0 < E; k0 += 64) {
    __builtin_prefetch(&xb[(size_t)(m0 + (lane & 15)) * E + k0 + 128], 0, 0);
    // prefetch k0+32 into buffer 1
#pragma unroll
    for (int s = 0; s < 2; ++s) a1[s] = load_a_frag(xb, E, m0 + s * 16, k0 + 32, lane);
#pragma unroll
    for (int t = 0; t < 4; ++t) b1[t] = load_b_frag(Wb, E, n0 + t * 16, k0 + 32, lane);
    // compute buffer 0
#pragma unroll
    for (int s = 0; s < 2; ++s)
#pragma unroll
      for (int t = 0; t < 4; ++t) acc[s][t] = WMMA_BF16(a0[s], b0[t], acc[s][t]);
    // prefetch k0+64 into buffer 0
    if (k0 + 64 < E) {
#pragma unroll
      for (int s = 0; s < 2; ++s) a0[s] = load_a_frag(xb, E, m0 + s * 16, k0 + 64, lane);
#pragma unroll
      for (int t = 0; t < 4; ++t) b0[t] = load_b_frag(Wb, E, n0 + t * 16, k0 + 64, lane);
    }
    // compute buffer 1
#pragma unroll
    for (int s = 0; s < 2; ++s)
#pragma unroll
      for (int t = 0; t < 4; ++t) acc[s][t] = WMMA_BF16(a1[s], b1[t], acc[s][t]);
  }

  // C/D layout -> LDS f32 tile. element r: (M = r + 8*(lane>=16), N = lane&15)
  const int half8 = (lane & 16) ? 8 : 0;
#pragma unroll
  for (int s = 0; s < 2; ++s)
#pragma unroll
    for (int t = 0; t < 4; ++t)
#pragma unroll
      for (int r = 0; r < 8; ++r)
        tile[mat][strip * 32 + s * 16 + r + half8][t * 16 + (lane & 15)] = acc[s][t][r];
  __syncthreads();

  // Epilogue: threads 0..191 each own one row (s position) of one matrix.
  {
    const int m2 = tid >> 6;  // 0=q 1=k 2=v
    const int row = tid & 63;
    const int sIdx = rb * 64 + row;
    float y[64];
#pragma unroll
    for (int d = 0; d < 64; ++d) y[d] = tile[m2][row][d];

    if (m2 < 2) {
      float mean = 0.f;
#pragma unroll
      for (int d = 0; d < 64; ++d) mean += y[d];
      mean *= (1.0f / 64.0f);
      float var = 0.f;
#pragma unroll
      for (int d = 0; d < 64; ++d) { float t0 = y[d] - mean; var += t0 * t0; }
      var *= (1.0f / 64.0f);
      const float inv = rsqrtf(var + 1e-5f);
      const float* g = (m2 == 0) ? qg : kg;
#pragma unroll
      for (int d = 0; d < 64; ++d) y[d] = (y[d] - mean) * inv * g[d];
      // RoPE (rotate-half)
      const float pos = (float)sIdx;
      float out[64];
#pragma unroll
      for (int d = 0; d < 32; ++d) {
        float invf = __expf(-(float)d * (9.210340372f / 32.0f));  // 10000^(-d/32)
        float ang = pos * invf;
        float c = __cosf(ang), sn = __sinf(ang);
        out[d]      = y[d] * c - y[d + 32] * sn;
        out[d + 32] = y[d + 32] * c + y[d] * sn;
      }
      unsigned short* dst = ((m2 == 0) ? qbuf : kbuf) + ((size_t)h * S + sIdx) * D;
#pragma unroll
      for (int d = 0; d < 64; ++d) dst[d] = f2bf(out[d]);
    } else {
#pragma unroll
      for (int d = 0; d < 64; ++d)
        vT[((size_t)h * 64 + d) * S + sIdx] = f2bf(y[d]);
    }
  }
}

// ---------------------------------------------------------------------------
// Kernel 2: flash attention over sliding window |i-j| <= 256.
// Grid: (S/64, H). Block: 128 threads = 4 waves; wave w -> 16-row q strip.
// K/V 64x64 tiles are staged into LDS (async global->LDS when available),
// DOUBLE-BUFFERED: block kb+1 streams in while kb is computed (ASYNCcnt
// producer/consumer). All WMMA fragments then come from LDS.
// ---------------------------------------------------------------------------
__device__ __forceinline__ void stage_kv(const unsigned short* __restrict__ kh,
                                         const unsigned short* __restrict__ vh,
                                         int ks, int tid,
                                         unsigned short (*kl)[72],
                                         unsigned short (*vl)[72]) {
  const int S = 8192;
  // K tile: rows j (64) x d (64) ; V tile: rows d (64) x s (64); 16B chunks.
#pragma unroll
  for (int i = 0; i < 4; ++i) {
    int id = i * 128 + tid;        // 0..511
    int rid = id >> 3;             // row 0..63
    int c = (id & 7) * 8;          // element offset 0..56
    ASYNC_CP(kh + (size_t)(ks + rid) * 64 + c, &kl[rid][c]);
  }
#pragma unroll
  for (int i = 0; i < 4; ++i) {
    int id = i * 128 + tid;
    int rid = id >> 3;
    int c = (id & 7) * 8;
    ASYNC_CP(vh + (size_t)rid * S + ks + c, &vl[rid][c]);
  }
}

__global__ void __launch_bounds__(128)
attn_window(const unsigned short* __restrict__ qbuf,
            const unsigned short* __restrict__ kbuf,
            const unsigned short* __restrict__ vT,
            unsigned short* __restrict__ obuf) {
  const int S = 8192, E = 1024, D = 64, NB = 128;
  __shared__ __align__(16) unsigned short klds[2][64][72];  // rows 144B (16B-aligned)
  __shared__ __align__(16) unsigned short vlds[2][64][72];
  __shared__ __align__(16) unsigned short plds[4][16][72];

  const int qb = blockIdx.x, h = blockIdx.y;
  const int tid = threadIdx.x;
  const int lane = tid & 31, wave = tid >> 5;
  const int qrow = qb * 64 + wave * 16;
  const unsigned short* qh = qbuf + (size_t)h * S * D;
  const unsigned short* kh = kbuf + (size_t)h * S * D;
  const unsigned short* vh = vT + (size_t)h * D * S;

  // Q fragments reused across all key blocks
  v16bf aq[2];
#pragma unroll
  for (int c = 0; c < 2; ++c) aq[c] = load_a_frag(qh, D, qrow, c * 32, lane);

  v8f o[4] = {};
  float mrow[8], lrow[8];
#pragma unroll
  for (int r = 0; r < 8; ++r) { mrow[r] = -3.0e38f; lrow[r] = 0.0f; }

  const float k2 = 0.18033688f;  // (1/sqrt(64)) * log2(e)
  const int half8 = (lane & 16) ? 8 : 0;
  const int N15 = lane & 15;
  const int kbb = (lane & 16) ? 16 : 0;
  const int kb0 = (qb - 4 < 0) ? 0 : qb - 4;
  const int kb1 = (qb + 4 > NB - 1) ? NB - 1 : qb + 4;

  int st = 0;
  stage_kv(kh, vh, kb0 * 64, tid, klds[0], vlds[0]);
  ASYNC_WAIT();
  __syncthreads();

  for (int kb = kb0; kb <= kb1; ++kb) {
    const int ks = kb * 64;
    // stream next block into the other stage while computing this one
    if (kb < kb1) stage_kv(kh, vh, ks + 64, tid, klds[st ^ 1], vlds[st ^ 1]);

    // ---- scores: S = Q K^T, B fragments from LDS K tile ----
    v8f sacc[4] = {};
#pragma unroll
    for (int c = 0; c < 2; ++c)
#pragma unroll
      for (int t = 0; t < 4; ++t) {
        v16bf b = lds_b_frag(&klds[st][t * 16 + N15][c * 32 + kbb]);
        sacc[t] = WMMA_BF16(aq[c], b, sacc[t]);
      }
    // ---- sliding-window mask ----
#pragma unroll
    for (int t = 0; t < 4; ++t) {
      int j = ks + t * 16 + N15;
#pragma unroll
      for (int r = 0; r < 8; ++r) {
        int i = qrow + r + half8;
        int dij = i - j; if (dij < 0) dij = -dij;
        if (dij > 256) sacc[t][r] = -3.0e38f;
      }
    }
    // ---- online softmax (row lives in one 16-lane half) ----
#pragma unroll
    for (int r = 0; r < 8; ++r) {
      float mx = fmaxf(fmaxf(sacc[0][r], sacc[1][r]), fmaxf(sacc[2][r], sacc[3][r]));
#pragma unroll
      for (int off = 1; off < 16; off <<= 1) mx = fmaxf(mx, __shfl_xor(mx, off, 32));
      float mnew = fmaxf(mrow[r], mx);
      float alpha = exp2f((mrow[r] - mnew) * k2);
      float rsum = 0.f;
#pragma unroll
      for (int t = 0; t < 4; ++t) {
        float p = exp2f((sacc[t][r] - mnew) * k2);
        sacc[t][r] = p;
        rsum += p;
      }
#pragma unroll
      for (int off = 1; off < 16; off <<= 1) rsum += __shfl_xor(rsum, off, 32);
      lrow[r] = lrow[r] * alpha + rsum;
      mrow[r] = mnew;
#pragma unroll
      for (int t = 0; t < 4; ++t) o[t][r] *= alpha;
    }
    // ---- P: C-layout -> LDS -> A-layout fragments (bf16) ----
#pragma unroll
    for (int t = 0; t < 4; ++t)
#pragma unroll
      for (int r = 0; r < 8; ++r)
        plds[wave][r + half8][t * 16 + N15] = f2bf(sacc[t][r]);
    // (LDS ops are in-order within a wave; compiler inserts s_wait_dscnt)
#pragma unroll
    for (int c = 0; c < 2; ++c) {
      const int kbh = (lane & 16) ? 16 : 0;
      Frag16 f;
      const unsigned short* pp = &plds[wave][N15][c * 32 + kbh];
      f.q[0] = *reinterpret_cast<const v4u*>(pp);
      f.q[1] = *reinterpret_cast<const v4u*>(pp + 8);
      // ---- O += P V : B[k][n] = vT[d = n][ks+k], from LDS V tile ----
#pragma unroll
      for (int t = 0; t < 4; ++t) {
        v16bf g = lds_b_frag(&vlds[st][t * 16 + N15][c * 32 + kbb]);
        o[t] = WMMA_BF16(f.v, g, o[t]);
      }
    }

    ASYNC_WAIT();      // next stage landed
    __syncthreads();   // everyone done reading current stage
    st ^= 1;
  }

  // ---- finalize: O / l, write bf16 [S][E] with e = h*64 + d ----
#pragma unroll
  for (int t = 0; t < 4; ++t)
#pragma unroll
    for (int r = 0; r < 8; ++r) {
      float val = o[t][r] / lrow[r];
      obuf[(size_t)(qrow + r + half8) * E + h * 64 + t * 16 + N15] = f2bf(val);
    }
}

// ---------------------------------------------------------------------------
// Kernel 3: output projection  out = O @ Wp^T + bp   (f32 output)
// Grid: (S/64, E/64). Block: 64 threads = 2 waves, each a 32x64 strip,
// software-pipelined like kernel 1.
// ---------------------------------------------------------------------------
__global__ void __launch_bounds__(64)
out_proj(const unsigned short* __restrict__ ob,
         const unsigned short* __restrict__ Wpb,
         const float* __restrict__ bp,
         float* __restrict__ out) {
  const int E = 1024;
  const int rb = blockIdx.x, nb = blockIdx.y;
  const int lane = threadIdx.x & 31, wave = threadIdx.x >> 5;
  const int m0 = rb * 64 + wave * 32;
  const int n0 = nb * 64;

  v8f acc[2][4] = {};
  v16bf a0[2], b0[4], a1[2], b1[4];
#pragma unroll
  for (int s = 0; s < 2; ++s) a0[s] = load_a_frag(ob, E, m0 + s * 16, 0, lane);
#pragma unroll
  for (int t = 0; t < 4; ++t) b0[t] = load_b_frag(Wpb, E, n0 + t * 16, 0, lane);

  for (int k0 = 0; k0 < E; k0 += 64) {
    __builtin_prefetch(&ob[(size_t)(m0 + (lane & 15)) * E + k0 + 128], 0, 0);
#pragma unroll
    for (int s = 0; s < 2; ++s) a1[s] = load_a_frag(ob, E, m0 + s * 16, k0 + 32, lane);
#pragma unroll
    for (int t = 0; t < 4; ++t) b1[t] = load_b_frag(Wpb, E, n0 + t * 16, k0 + 32, lane);
#pragma unroll
    for (int s = 0; s < 2; ++s)
#pragma unroll
      for (int t = 0; t < 4; ++t) acc[s][t] = WMMA_BF16(a0[s], b0[t], acc[s][t]);
    if (k0 + 64 < E) {
#pragma unroll
      for (int s = 0; s < 2; ++s) a0[s] = load_a_frag(ob, E, m0 + s * 16, k0 + 64, lane);
#pragma unroll
      for (int t = 0; t < 4; ++t) b0[t] = load_b_frag(Wpb, E, n0 + t * 16, k0 + 64, lane);
    }
#pragma unroll
    for (int s = 0; s < 2; ++s)
#pragma unroll
      for (int t = 0; t < 4; ++t) acc[s][t] = WMMA_BF16(a1[s], b1[t], acc[s][t]);
  }

  const int half8 = (lane & 16) ? 8 : 0;
#pragma unroll
  for (int s = 0; s < 2; ++s)
#pragma unroll
    for (int t = 0; t < 4; ++t) {
      int n = n0 + t * 16 + (lane & 15);
      float bias = bp[n];
#pragma unroll
      for (int r = 0; r < 8; ++r)
        out[(size_t)(m0 + s * 16 + r + half8) * E + n] = acc[s][t][r] + bias;
    }
}

// ---------------------------------------------------------------------------
extern "C" void kernel_launch(void* const* d_in, const int* in_sizes, int n_in,
                              void* d_out, int out_size, void* d_ws, size_t ws_size,
                              hipStream_t stream) {
  const int S = 8192, E = 1024, H = 16;
  const float* x  = (const float*)d_in[0];
  const float* Wq = (const float*)d_in[1];
  const float* Wk = (const float*)d_in[2];
  const float* Wv = (const float*)d_in[3];
  const float* Wp = (const float*)d_in[4];
  const float* bp = (const float*)d_in[5];
  const float* qg = (const float*)d_in[6];
  const float* kg = (const float*)d_in[7];
  (void)in_sizes; (void)n_in; (void)out_size; (void)ws_size;

  const size_t SE = (size_t)S * E;
  const size_t EE = (size_t)E * E;
  unsigned short* ws  = (unsigned short*)d_ws;
  unsigned short* xb  = ws;            // x bf16          [S][E]
  unsigned short* Wqb = xb + SE;       // weights bf16    [E][E]
  unsigned short* Wkb = Wqb + EE;
  unsigned short* Wvb = Wkb + EE;
  unsigned short* Wpb = Wvb + EE;
  unsigned short* qb_ = Wpb + EE;      // q (LN+RoPE)     [H][S][D]
  unsigned short* kb_ = qb_ + SE;      // k (LN+RoPE)     [H][S][D]
  unsigned short* vTb = kb_ + SE;      // v transposed    [H][D][S]
  unsigned short* ob_ = vTb + SE;      // attention out   [S][E]

  const int th = 256;
  cvt_f32_bf16<<<(unsigned)((SE + th - 1) / th), th, 0, stream>>>(x,  xb,  (int)SE);
  cvt_f32_bf16<<<(unsigned)((EE + th - 1) / th), th, 0, stream>>>(Wq, Wqb, (int)EE);
  cvt_f32_bf16<<<(unsigned)((EE + th - 1) / th), th, 0, stream>>>(Wk, Wkb, (int)EE);
  cvt_f32_bf16<<<(unsigned)((EE + th - 1) / th), th, 0, stream>>>(Wv, Wvb, (int)EE);
  cvt_f32_bf16<<<(unsigned)((EE + th - 1) / th), th, 0, stream>>>(Wp, Wpb, (int)EE);

  qkv_ln_rope<<<dim3(S / 64, H), 192, 0, stream>>>(xb, Wqb, Wkb, Wvb, qg, kg,
                                                   qb_, kb_, vTb);
  attn_window<<<dim3(S / 64, H), 128, 0, stream>>>(qb_, kb_, vTb, ob_);
  out_proj<<<dim3(S / 64, E / 64), 64, 0, stream>>>(ob_, Wpb, bp, (float*)d_out);
}
